// PatchInpainting_13572096655741
// MI455X (gfx1250) — matrix operand order repaired and checked by reading, not compiled, not run on hardware
//
#include <hip/hip_runtime.h>
#include <hip/hip_bf16.h>

typedef __attribute__((ext_vector_type(16))) _Float16 v16h;
typedef __attribute__((ext_vector_type(8)))  float    v8f;

#define B_     2
#define CH_    3
#define H_     512
#define KS_    8
#define G_     64
#define N_     4096
#define HEADS_ 4
#define DH_    80
#define DPAD_  96
#define TOPK_  64
#define VPS_   16
#define PAD_   4
#define DV_    192
#define VTOK_  768
#define NEG_   -1000000000.0f
#define INV_SQRT_TEMP 3.16227766016838f   // sqrt(1/TEMP), folded into descriptors

// ---------------------------------------------------------------------------
// Kernel 1: build L2-normalized per-head descriptors (f32 math, f16 store),
// K padded 80 -> 96 with zeros. Layout: desc[(b*4+h)*N + q][96]
// ---------------------------------------------------------------------------
__global__ void k_desc(const float* __restrict__ img, const float* __restrict__ feat,
                       const int* __restrict__ mask, _Float16* __restrict__ desc) {
    int tok = blockIdx.x;                  // b*N + q
    int b = tok >> 12, q = tok & 4095;
    int gy = q >> 6, gx = q & 63;
    int t = threadIdx.x;                   // 128 threads
    __shared__ float red[128];
    int m = mask[tok];
    for (int h = 0; h < HEADS_; ++h) {
        float v = 0.f;
        if (t < DH_) {
            int D = h * DH_ + t;
            if (D < 192) {                 // masked RGB patch part
                int c = D >> 6, r = D & 63, py = r >> 3, px = r & 7;
                v = m ? 0.f : img[((b * CH_ + c) * H_ + gy * KS_ + py) * H_ + gx * KS_ + px];
            } else {                       // feature part
                v = feat[(b * 128 + (D - 192)) * N_ + q];
            }
        }
        red[t] = v * v;
        __syncthreads();
        for (int s = 64; s > 0; s >>= 1) {
            if (t < s) red[t] += red[t + s];
            __syncthreads();
        }
        float scale = INV_SQRT_TEMP / (sqrtf(red[0]) + 1e-6f);
        __syncthreads();                   // red[] reused next head
        if (t < DPAD_) {
            _Float16 o = (t < DH_) ? (_Float16)(v * scale) : (_Float16)0.f;
            desc[((long)(b * HEADS_ + h) * N_ + q) * DPAD_ + t] = o;
        }
    }
}

// ---------------------------------------------------------------------------
// Kernel 2: swizzle descriptors into wave32 WMMA fragment order (ISA 7.12.2)
// so the attention kernel does contiguous 32B/lane loads.
//   fragA/fragB element layout: [((bh*256+tile)*3+kc)*32 + lane][16 halves]
// ---------------------------------------------------------------------------
__global__ void k_frag(const _Float16* __restrict__ desc,
                       _Float16* __restrict__ fragA, _Float16* __restrict__ fragB) {
    int idx = blockIdx.x;                  // ((b*H+h)*256 + tile)*3 + kc
    int kc = idx % 3;
    int tile = (idx / 3) & 255;
    int bh = idx / 768;
    int t = threadIdx.x;                   // 64 threads
    int lane = t & 31;
    const _Float16* dbase = desc + (long)bh * N_ * DPAD_;
    long fo = ((long)idx * 32 + lane) * 16;
    _Float16 tmp[16] __attribute__((aligned(32)));
    if (t < 32) {
        // A 16x32 f16: lanes 0-15 row m, K {0..7,16..23}; lanes 16-31 K {8..15,24..31}
        int mrow = tile * 16 + (lane & 15);
        int kb = (lane < 16) ? 0 : 8;
        for (int i = 0; i < 16; ++i) {
            int K = (i < 8) ? (i + kb) : (i + 8 + kb);
            tmp[i] = dbase[(long)mrow * DPAD_ + kc * 32 + K];
        }
        *(v16h*)(fragA + fo) = *(v16h*)tmp;
    } else {
        // B 32x16 f16: lanes 0-15 col n, K {0..7,16..23}; lanes 16-31 col n, K {8..15,24..31}
        int ncol = tile * 16 + (lane & 15);
        int koff = (lane < 16) ? 0 : 8;
        for (int i = 0; i < 16; ++i)
            tmp[i] = dbase[(long)ncol * DPAD_ + kc * 32 + ((i < 8) ? (i + koff) : (i + 8 + koff))];
        *(v16h*)(fragB + fo) = *(v16h*)tmp;
    }
}

// ---------------------------------------------------------------------------
// Kernel 3: unfold value tokens (16x16x3 windows, stride 8, pad 4, UNmasked img)
// ---------------------------------------------------------------------------
__global__ void k_val(const float* __restrict__ img, float* __restrict__ val) {
    int tok = blockIdx.x;                  // b*N + q
    int b = tok >> 12, q = tok & 4095;
    int gy = q >> 6, gx = q & 63;
    int t = threadIdx.x;                   // 256 threads, 3 elems each
    int y0 = gy * KS_ - PAD_, x0 = gx * KS_ - PAD_;
    for (int s = 0; s < 3; ++s) {
        int e = t + 256 * s;
        int c = e >> 8, r = e & 255, iy = r >> 4, ix = r & 15;
        int y = y0 + iy, x = x0 + ix;
        float v = 0.f;
        if ((unsigned)y < H_ && (unsigned)x < H_)
            v = img[((b * CH_ + c) * H_ + y) * H_ + x];
        val[(long)tok * VTOK_ + e] = v;
    }
}

// ---------------------------------------------------------------------------
// Kernel 4: fused WMMA scores + key mask + per-query top-64 + softmax.
// One WG = (b, h, 16 queries). 512 threads = 16 waves. LDS ~280KB => exactly
// 1 WG/WGP (4 waves/SIMD32), so latency hiding comes from in-wave ILP:
// each wave processes TWO key tiles per iteration with independent WMMA
// accumulator chains (fills the f16 WMMA->WMMA hazard slots, ISA 7.12.1).
// Key-mask table is staged into LDS via the gfx1250 async-to-LDS path.
// ---------------------------------------------------------------------------
__global__ __launch_bounds__(512) void k_attn(const _Float16* __restrict__ fragA,
                                              const _Float16* __restrict__ fragB,
                                              const int* __restrict__ mask,
                                              float* __restrict__ attw,
                                              int* __restrict__ atti) {
    extern __shared__ char smem[];
    float* sc = (float*)smem;              // 16 * 4096 score stripe
    float* kb = sc + 16 * N_;              // 4096 key bias (0 or NEG)
    float* tv = kb + N_;                   // 16 * 64 top values
    int*   ti = (int*)(tv + 16 * TOPK_);   // 16 * 64 top indices

    int qt = blockIdx.x & 255;
    int bh = blockIdx.x >> 8;              // b*HEADS + h
    int b  = bh >> 2;
    int tid = threadIdx.x;
    int wave = tid >> 5, lane = tid & 31;

    // ---- async-stage key mask words into the kb LDS region (ASYNCcnt path).
    // No static LDS in this kernel => dynamic-LDS byte offset == LDS offset.
    for (int k = tid; k < N_; k += 512) {
        unsigned loff = (unsigned)((16 * N_ + k) * 4);
        unsigned long long ga = (unsigned long long)(const void*)&mask[b * N_ + k];
        asm volatile("global_load_async_to_lds_b32 %0, %1, off"
                     :: "v"(loff), "v"(ga) : "memory");
    }
    asm volatile("s_wait_asynccnt 0x0" ::: "memory");
    // convert int -> bias in place (each thread converts the words its own
    // wave copied, so the per-wave ASYNCcnt wait above is sufficient)
    for (int k = tid; k < N_; k += 512) {
        int mv = ((volatile int*)kb)[k];
        ((volatile float*)kb)[k] = (mv == 0) ? 0.f : NEG_;
    }
    __syncthreads();

    // A fragments for this query tile (shared by all 16 waves), 3 K-chunks
    long abase = ((long)bh * 256 + qt) * 3 * 512;
    v16h a0 = *(const v16h*)(fragA + abase + 0 * 512 + lane * 16);
    v16h a1 = *(const v16h*)(fragA + abase + 1 * 512 + lane * 16);
    v16h a2 = *(const v16h*)(fragA + abase + 2 * 512 + lane * 16);

    // each wave sweeps 16 of 256 key tiles, two per iteration (dual chains)
    for (int i = 0; i < 8; ++i) {
        int kt0 = wave + 32 * i;
        int kt1 = kt0 + 16;
        long bb0 = ((long)bh * 256 + kt0) * 3 * 512;
        long bb1 = ((long)bh * 256 + kt1) * 3 * 512;
        v16h b00 = *(const v16h*)(fragB + bb0 + 0 * 512 + lane * 16);
        v16h b01 = *(const v16h*)(fragB + bb0 + 1 * 512 + lane * 16);
        v16h b02 = *(const v16h*)(fragB + bb0 + 2 * 512 + lane * 16);
        v16h b10 = *(const v16h*)(fragB + bb1 + 0 * 512 + lane * 16);
        v16h b11 = *(const v16h*)(fragB + bb1 + 1 * 512 + lane * 16);
        v16h b12 = *(const v16h*)(fragB + bb1 + 2 * 512 + lane * 16);
        v8f c0 = {}, c1 = {};
        c0 = __builtin_amdgcn_wmma_f32_16x16x32_f16(false, a0, false, b00, (short)0, c0, false, false);
        c1 = __builtin_amdgcn_wmma_f32_16x16x32_f16(false, a0, false, b10, (short)0, c1, false, false);
        c0 = __builtin_amdgcn_wmma_f32_16x16x32_f16(false, a1, false, b01, (short)0, c0, false, false);
        c1 = __builtin_amdgcn_wmma_f32_16x16x32_f16(false, a1, false, b11, (short)0, c1, false, false);
        c0 = __builtin_amdgcn_wmma_f32_16x16x32_f16(false, a2, false, b02, (short)0, c0, false, false);
        c1 = __builtin_amdgcn_wmma_f32_16x16x32_f16(false, a2, false, b12, (short)0, c1, false, false);
        // C layout: lane%16 = N col, VGPR r -> row r (+8 for lanes 16-31)
        int col0 = kt0 * 16 + (lane & 15);
        int col1 = kt1 * 16 + (lane & 15);
        int moff = (lane < 16) ? 0 : 8;
        float bias0 = kb[col0];
        float bias1 = kb[col1];
#pragma unroll
        for (int r = 0; r < 8; ++r) {
            sc[(r + moff) * N_ + col0] = c0[r] + bias0;
            sc[(r + moff) * N_ + col1] = c1[r] + bias1;
        }
    }
    __syncthreads();

    // ---- top-64: wave w owns query row w. Per-lane cached max over its
    // 128-element strided subset; only the owner lane rescans per round. ----
    float* row = sc + wave * N_;
    float lbest = 2.f * NEG_; int lidx = lane;
    for (int j = lane; j < N_; j += 32) {
        float v = row[j];
        if (v > lbest) { lbest = v; lidx = j; }
    }
    for (int it = 0; it < TOPK_; ++it) {
        float gbest = lbest; int gidx = lidx;
#pragma unroll
        for (int off = 16; off > 0; off >>= 1) {
            float ov = __shfl_xor(gbest, off, 32);
            int   oi = __shfl_xor(gidx, off, 32);
            if (ov > gbest || (ov == gbest && oi < gidx)) { gbest = ov; gidx = oi; }
        }
        if (lane == 0) { tv[wave * TOPK_ + it] = gbest; ti[wave * TOPK_ + it] = gidx; }
        if ((gidx & 31) == lane) {         // owner lane invalidates + rescans subset
            row[gidx] = 2.f * NEG_;
            lbest = 2.f * NEG_; lidx = lane;
            for (int j = lane; j < N_; j += 32) {
                float v = row[j];
                if (v > lbest) { lbest = v; lidx = j; }
            }
        }
    }

    // ---- softmax over the 64 selected scores (2 per lane) ----
    float m0 = tv[wave * TOPK_ + lane];
    float m1 = tv[wave * TOPK_ + lane + 32];
    float mx = fmaxf(m0, m1);
#pragma unroll
    for (int off = 16; off > 0; off >>= 1) mx = fmaxf(mx, __shfl_xor(mx, off, 32));
    float e0 = __expf(m0 - mx), e1 = __expf(m1 - mx);
    float ss = e0 + e1;
#pragma unroll
    for (int off = 16; off > 0; off >>= 1) ss += __shfl_xor(ss, off, 32);
    float inv = 1.f / ss;
    long obase = ((long)bh * N_ + qt * 16 + wave) * TOPK_;
    attw[obase + lane]      = e0 * inv;
    attw[obase + lane + 32] = e1 * inv;
    atti[obase + lane]      = ti[wave * TOPK_ + lane];
    atti[obase + lane + 32] = ti[wave * TOPK_ + lane + 32];
}

// ---------------------------------------------------------------------------
// Kernel 5: attention-weighted gather of value tokens. One WG per (b, q),
// 256 threads x 3 output floats; val rows hit in L2 (50MB << 192MB).
// ---------------------------------------------------------------------------
__global__ void k_gather(const float* __restrict__ val, const float* __restrict__ attw,
                         const int* __restrict__ atti, float* __restrict__ outtok) {
    int tok = blockIdx.x;                  // b*N + q
    int b = tok >> 12, q = tok & 4095;
    int t = threadIdx.x;                   // 256 = HEADS*TOPK
    __shared__ float sw[HEADS_ * TOPK_];
    __shared__ int   si[HEADS_ * TOPK_];
    {
        int h = t >> 6, jj = t & 63;
        long base = ((long)(b * HEADS_ + h) * N_ + q) * TOPK_ + jj;
        sw[t] = attw[base];
        si[t] = atti[base];
    }
    __syncthreads();
    int e0 = t, e1 = t + 256, e2 = t + 512;
    int h0 = e0 / DV_, h1 = e1 / DV_, h2 = e2 / DV_;
    const float* vb = val + (long)b * N_ * VTOK_;
    float acc0 = 0.f, acc1 = 0.f, acc2 = 0.f;
    for (int j = 0; j < TOPK_; ++j) {
        float w0 = sw[h0 * TOPK_ + j]; int i0 = si[h0 * TOPK_ + j];
        float w1 = sw[h1 * TOPK_ + j]; int i1 = si[h1 * TOPK_ + j];
        float w2 = sw[h2 * TOPK_ + j]; int i2 = si[h2 * TOPK_ + j];
        acc0 += w0 * vb[(long)i0 * VTOK_ + e0];
        acc1 += w1 * vb[(long)i1 * VTOK_ + e1];
        acc2 += w2 * vb[(long)i2 * VTOK_ + e2];
        if (j + 1 < TOPK_)                 // global_prefetch_b8 of next key row
            __builtin_prefetch(&vb[(long)si[h0 * TOPK_ + j + 1] * VTOK_ + e0], 0, 1);
    }
    long ob = (long)tok * VTOK_;
    outtok[ob + e0] = acc0; outtok[ob + e1] = acc1; outtok[ob + e2] = acc2;
}

// ---------------------------------------------------------------------------
// Kernel 6: fold (overlap-add gather form: <=4 contributors/pixel) + composite
// ---------------------------------------------------------------------------
__global__ void k_fold(const float* __restrict__ outtok, const int* __restrict__ mask,
                       const float* __restrict__ img, float* __restrict__ out) {
    long p = (long)blockIdx.x * 256 + threadIdx.x;   // over B*CH*H*H
    int x = (int)(p & 511);
    int y = (int)((p >> 9) & 511);
    int c = (int)((p >> 18) % 3);
    int b = (int)(p / (3L << 18));
    int yp = y + PAD_, xp = x + PAD_;
    int gylo = (yp >= 15) ? ((yp - 15 + 7) >> 3) : 0;
    int gyhi = min(63, yp >> 3);
    int gxlo = (xp >= 15) ? ((xp - 15 + 7) >> 3) : 0;
    int gxhi = min(63, xp >> 3);
    float num = 0.f, den = 0.f;
    for (int gy = gylo; gy <= gyhi; ++gy)
        for (int gx = gxlo; gx <= gxhi; ++gx) {
            int iy = yp - gy * 8, ix = xp - gx * 8;
            num += outtok[((long)b * N_ + gy * 64 + gx) * VTOK_ + c * 256 + iy * 16 + ix];
            den += 1.f;
        }
    float recon = num / fmaxf(den, 1e-6f);
    int m = mask[b * N_ + (y >> 3) * 64 + (x >> 3)];
    out[p] = m ? recon : img[p];
}

// ---------------------------------------------------------------------------
extern "C" void kernel_launch(void* const* d_in, const int* in_sizes, int n_in,
                              void* d_out, int out_size, void* d_ws, size_t ws_size,
                              hipStream_t stream) {
    (void)in_sizes; (void)n_in; (void)out_size; (void)ws_size;
    const float* img  = (const float*)d_in[0];   // (2,3,512,512)
    const float* feat = (const float*)d_in[1];   // (2,128,64,64)
    const int*   mask = (const int*)d_in[2];     // (2,1,64,64)
    float* out = (float*)d_out;                  // (2,3,512,512)

    char* ws = (char*)d_ws;
    auto alloc = [&](size_t bytes) -> char* {
        char* p = ws; ws += (bytes + 255) & ~(size_t)255; return p;
    };
    _Float16* desc  = (_Float16*)alloc((size_t)B_ * HEADS_ * N_ * DPAD_ * 2); //  6.3MB
    _Float16* fragA = (_Float16*)alloc((size_t)B_ * HEADS_ * N_ * DPAD_ * 2); //  6.3MB
    _Float16* fragB = (_Float16*)alloc((size_t)B_ * HEADS_ * N_ * DPAD_ * 2); //  6.3MB
    float* val    = (float*)alloc((size_t)B_ * N_ * VTOK_ * 4);               // 25.2MB
    float* attw   = (float*)alloc((size_t)B_ * HEADS_ * N_ * TOPK_ * 4);      //  8.4MB
    int*   atti   = (int*)  alloc((size_t)B_ * HEADS_ * N_ * TOPK_ * 4);      //  8.4MB
    float* outtok = (float*)alloc((size_t)B_ * N_ * VTOK_ * 4);               // 25.2MB

    k_desc<<<B_ * N_, 128, 0, stream>>>(img, feat, mask, desc);
    k_frag<<<B_ * HEADS_ * 256 * 3, 64, 0, stream>>>(desc, fragA, fragB);
    k_val<<<B_ * N_, 256, 0, stream>>>(img, val);

    size_t smem = (size_t)(16 * N_ + N_ + 16 * TOPK_) * 4 + (size_t)16 * TOPK_ * 4; // 286,720B
    k_attn<<<B_ * HEADS_ * 256, 512, smem, stream>>>(fragA, fragB, mask, attw, atti);

    k_gather<<<B_ * N_, 256, 0, stream>>>(val, attw, atti, outtok);
    k_fold<<<(B_ * CH_ * H_ * H_) / 256, 256, 0, stream>>>(outtok, mask, img, out);
}
